// OptVQ_31885837205546
// MI455X (gfx1250) — matrix operand order repaired and checked by reading, not compiled, not run on hardware
//
#include <hip/hip_runtime.h>
#include <hip/hip_bf16.h>

// ---------------------------------------------------------------------------
// OptVQ forward for MI455X (gfx1250, wave32, WMMA + async-to-LDS staging).
//  z:(32,256,32,32) f32, embedding:(1,1024,256), W:(256,256), b:(256)
//  N=32768 points x C=256 dims, K=1024 codes.
//  d = cdist(z_flat, emb)  via bf16 WMMA GEMM (f32 accum) + norm trick
//  Sinkhorn factored as Q = E * (r outer c); argmax_k = argmax_k E[n][k]*r[k]
// ---------------------------------------------------------------------------

typedef __attribute__((ext_vector_type(16))) __bf16 v16bf;
typedef __attribute__((ext_vector_type(8)))  float  v8f;

#define NPTS   32768
#define KCODES 1024
#define DDIM   256
#define NK_D   33554432.0   // NPTS*KCODES
#define EPSI   1e-8f

struct Scal {            // scalar accumulators in workspace
    double sum;          // sum of d
    double sumsq;        // sum of d^2
    double T;            // sum of E
    double mse;          // sum of (zq - zc)^2
    unsigned minbits;    // min of d (float bits, d>=0 so monotone)
    float alpha, beta;   // E = exp(alpha*d + beta)
};

// ---------------- prep kernels ----------------

__global__ void k0_init(Scal* sc) {
    if (threadIdx.x == 0) {
        sc->sum = 0.0; sc->sumsq = 0.0; sc->T = 0.0; sc->mse = 0.0;
        sc->minbits = 0x7F800000u; // +inf
    }
}

// emb[k][d] = sum_j embedding[k][j] * W[d][j] + b[d]     (block=k, thread=d)
__global__ __launch_bounds__(256) void k1_linear(const float* __restrict__ E0,
        const float* __restrict__ W, const float* __restrict__ bias,
        float* __restrict__ emb) {
    int k = blockIdx.x, d = threadIdx.x;
    const float* er = E0 + (size_t)k * DDIM;
    const float* wr = W  + (size_t)d * DDIM;
    float s = bias[d];
    #pragma unroll 8
    for (int j = 0; j < DDIM; ++j) s = fmaf(er[j], wr[j], s);
    emb[(size_t)k * DDIM + d] = s;
}

// embB (bf16) + y2[k] = ||emb_k||^2
__global__ __launch_bounds__(256) void k2_embprep(const float* __restrict__ emb,
        __bf16* __restrict__ embB, float* __restrict__ y2) {
    __shared__ float red[256];
    int k = blockIdx.x, d = threadIdx.x;
    float v = emb[(size_t)k * DDIM + d];
    embB[(size_t)k * DDIM + d] = (__bf16)v;
    red[d] = v * v; __syncthreads();
    for (int s = 128; s > 0; s >>= 1) { if (d < s) red[d] += red[d + s]; __syncthreads(); }
    if (d == 0) y2[k] = red[0];
}

// zB[n][c] (bf16, NHWC gather) + x2[n] = ||z_n||^2
__global__ __launch_bounds__(256) void k3_zprep(const float* __restrict__ z,
        __bf16* __restrict__ zB, float* __restrict__ x2) {
    __shared__ float red[256];
    int n = blockIdx.x, c = threadIdx.x;
    int b = n >> 10, h = (n >> 5) & 31, w = n & 31;
    float v = z[(size_t)b * 262144 + (size_t)c * 1024 + h * 32 + w];
    zB[(size_t)n * DDIM + c] = (__bf16)v;
    red[c] = v * v; __syncthreads();
    for (int s = 128; s > 0; s >>= 1) { if (c < s) red[c] += red[c + s]; __syncthreads(); }
    if (c == 0) x2[n] = red[0];
}

// ---------------- WMMA GEMM + distance + stats ----------------
// block = 256 thr (8 waves); block tile 128 rows x 128 codes; K-step 32 x 8.
// Wave tile 32x64 (2 A-frags x 4 B-frags -> 8 WMMA per K-step, straight-line).
// Staging: double-buffered LDS filled by GLOBAL_LOAD_ASYNC_TO_LDS_B128.
#define LDA 40  // padded LDS row stride (bf16) -> 80B, bank-spread

__global__ __launch_bounds__(256) void k4_gemm(const __bf16* __restrict__ zB,
        const __bf16* __restrict__ embB, const float* __restrict__ x2,
        const float* __restrict__ y2, float* __restrict__ dbuf, Scal* sc) {
    __shared__ alignas(16) __bf16 sA[2][128][LDA];
    __shared__ alignas(16) __bf16 sB[2][128][LDA];
    __shared__ float red[256];
    const int tid  = threadIdx.x;
    const int lane = tid & 31;
    const int wave = tid >> 5;             // 0..7
    const int wr   = (wave & 3) << 5;      // wave row offset within tile: 0/32/64/96
    const int wc   = (wave >> 2) << 6;     // wave col offset within tile: 0/64
    const int rowBase = blockIdx.x * 128;  // n
    const int colBase = blockIdx.y * 128;  // k

    v8f acc[2][4] = {};

    const int lrow = tid >> 1;             // 0..127 (staging row)
    const int lcol = (tid & 1) << 4;       // 0 or 16 (staging bf16 col)
    const int asel = (lane >> 4) << 3;     // A K-chunk base elem: 0 or 8
    const int bsel = (lane >> 4) << 3;     // B K-chunk base elem: 0 or 8
    const int ln15 = lane & 15;

    // async stage of one 128x32 A chunk + 128x32 B chunk into LDS buffer `buf`.
    // INST_OFFSET is added to BOTH lds and global addresses (ISA 15.18), so one
    // base covers both 16B beats.
    auto stage_async = [&](int buf, int kk) {
        uint64_t ga = (uint64_t)(zB   + (size_t)(rowBase + lrow) * DDIM + kk * 32 + lcol);
        uint64_t gb = (uint64_t)(embB + (size_t)(colBase + lrow) * DDIM + kk * 32 + lcol);
        unsigned la = (unsigned)(uintptr_t)&sA[buf][lrow][lcol];
        unsigned lb = (unsigned)(uintptr_t)&sB[buf][lrow][lcol];
        asm volatile("global_load_async_to_lds_b128 %0, %1, off"
                     :: "v"(la), "v"(ga) : "memory");
        asm volatile("global_load_async_to_lds_b128 %0, %1, off offset:16"
                     :: "v"(la), "v"(ga) : "memory");
        asm volatile("global_load_async_to_lds_b128 %0, %1, off"
                     :: "v"(lb), "v"(gb) : "memory");
        asm volatile("global_load_async_to_lds_b128 %0, %1, off offset:16"
                     :: "v"(lb), "v"(gb) : "memory");
    };

    union Frag { v16bf v; uint4 q[2]; };

    stage_async(0, 0);
    asm volatile("s_wait_asynccnt 0x0" ::: "memory");
    __syncthreads();

    for (int kk = 0; kk < 8; ++kk) {
        const int cur = kk & 1;
        if (kk < 7) stage_async(cur ^ 1, kk + 1);   // overlap HBM->LDS with WMMA

        // A fragments (16x32): lane<16 -> K{0..7,16..23}, lane>=16 -> K{8..15,24..31}
        Frag a0, a1;
        a0.q[0] = *(const uint4*)&sA[cur][wr + ln15][asel];
        a0.q[1] = *(const uint4*)&sA[cur][wr + ln15][asel + 16];
        a1.q[0] = *(const uint4*)&sA[cur][wr + 16 + ln15][asel];
        a1.q[1] = *(const uint4*)&sA[cur][wr + 16 + ln15][asel + 16];
        // B fragments (32x16): col N = lane&15, lane<16 -> K{0..7,16..23}, lane>=16 -> K{8..15,24..31}.
        // All four loaded into distinct SSA values -> no loop-carried copies.
        Frag b0, b1, b2, b3;
        b0.q[0] = *(const uint4*)&sB[cur][wc +  0 + ln15][bsel];
        b0.q[1] = *(const uint4*)&sB[cur][wc +  0 + ln15][bsel + 16];
        b1.q[0] = *(const uint4*)&sB[cur][wc + 16 + ln15][bsel];
        b1.q[1] = *(const uint4*)&sB[cur][wc + 16 + ln15][bsel + 16];
        b2.q[0] = *(const uint4*)&sB[cur][wc + 32 + ln15][bsel];
        b2.q[1] = *(const uint4*)&sB[cur][wc + 32 + ln15][bsel + 16];
        b3.q[0] = *(const uint4*)&sB[cur][wc + 48 + ln15][bsel];
        b3.q[1] = *(const uint4*)&sB[cur][wc + 48 + ln15][bsel + 16];

        acc[0][0] = __builtin_amdgcn_wmma_f32_16x16x32_bf16(
            false, a0.v, false, b0.v, (short)0, acc[0][0], false, false);
        acc[1][0] = __builtin_amdgcn_wmma_f32_16x16x32_bf16(
            false, a1.v, false, b0.v, (short)0, acc[1][0], false, false);
        acc[0][1] = __builtin_amdgcn_wmma_f32_16x16x32_bf16(
            false, a0.v, false, b1.v, (short)0, acc[0][1], false, false);
        acc[1][1] = __builtin_amdgcn_wmma_f32_16x16x32_bf16(
            false, a1.v, false, b1.v, (short)0, acc[1][1], false, false);
        acc[0][2] = __builtin_amdgcn_wmma_f32_16x16x32_bf16(
            false, a0.v, false, b2.v, (short)0, acc[0][2], false, false);
        acc[1][2] = __builtin_amdgcn_wmma_f32_16x16x32_bf16(
            false, a1.v, false, b2.v, (short)0, acc[1][2], false, false);
        acc[0][3] = __builtin_amdgcn_wmma_f32_16x16x32_bf16(
            false, a0.v, false, b3.v, (short)0, acc[0][3], false, false);
        acc[1][3] = __builtin_amdgcn_wmma_f32_16x16x32_bf16(
            false, a1.v, false, b3.v, (short)0, acc[1][3], false, false);

        if (kk < 7) asm volatile("s_wait_asynccnt 0x0" ::: "memory");
        __syncthreads();
    }

    // epilogue: d = sqrt(max(x2 + y2 - 2*dot, 0)); C/D: M=r+8*(lane>=16), N=lane&15
    const int M0 = (lane >> 4) << 3;
    float lsum = 0.f, lsq = 0.f, lmin = 3.4e38f;
    #pragma unroll
    for (int mi = 0; mi < 2; ++mi) {
        float xv[8];
        #pragma unroll
        for (int r = 0; r < 8; ++r) xv[r] = x2[rowBase + wr + (mi << 4) + M0 + r];
        #pragma unroll
        for (int t = 0; t < 4; ++t) {
            const int gCol = colBase + wc + (t << 4) + ln15;
            const float yv = y2[gCol];
            #pragma unroll
            for (int r = 0; r < 8; ++r) {
                const int gRow = rowBase + wr + (mi << 4) + M0 + r;
                float f = xv[r] + yv - 2.0f * acc[mi][t][r];
                f = sqrtf(fmaxf(f, 0.0f));
                dbuf[(size_t)gRow * KCODES + gCol] = f;
                lsum += f; lsq += f * f; lmin = fminf(lmin, f);
            }
        }
    }
    // block reductions -> f64/min atomics
    red[tid] = lsum; __syncthreads();
    for (int s = 128; s > 0; s >>= 1) { if (tid < s) red[tid] += red[tid + s]; __syncthreads(); }
    if (tid == 0) atomicAdd(&sc->sum, (double)red[0]);
    __syncthreads();
    red[tid] = lsq; __syncthreads();
    for (int s = 128; s > 0; s >>= 1) { if (tid < s) red[tid] += red[tid + s]; __syncthreads(); }
    if (tid == 0) atomicAdd(&sc->sumsq, (double)red[0]);
    __syncthreads();
    red[tid] = lmin; __syncthreads();
    for (int s = 128; s > 0; s >>= 1) { if (tid < s) red[tid] = fminf(red[tid], red[tid + s]); __syncthreads(); }
    if (tid == 0) atomicMin(&sc->minbits, __float_as_uint(red[0]));
}

// ---------------- sinkhorn ----------------

__global__ void k5_coeff(Scal* sc) {
    double mean = sc->sum / NK_D;
    double var  = (sc->sumsq - NK_D * mean * mean) / (NK_D - 1.0);
    if (var < 0.0) var = 0.0;
    double denom = sqrt(var) + 1e-8;
    float dmin = __uint_as_float(sc->minbits);
    // E = exp(-10*(d - dmin)/denom) = exp(alpha*d + beta)
    sc->alpha = (float)(-10.0 / denom);
    sc->beta  = (float)( 10.0 * (double)dmin / denom);
}

// in-place d -> E, accumulate T = sum(E). 4 elems/thread, strided.
__global__ __launch_bounds__(256) void k6_exp(float* __restrict__ E, Scal* sc) {
    __shared__ float red[256];
    const float a = sc->alpha, b = sc->beta;
    const size_t base = (size_t)blockIdx.x * 1024 + threadIdx.x;
    float s = 0.f;
    #pragma unroll
    for (int j = 0; j < 4; ++j) {
        size_t ix = base + (size_t)j * 256;
        float e = __expf(fmaf(a, E[ix], b));
        E[ix] = e; s += e;
    }
    red[threadIdx.x] = s; __syncthreads();
    for (int st = 128; st > 0; st >>= 1) { if (threadIdx.x < st) red[threadIdx.x] += red[threadIdx.x + st]; __syncthreads(); }
    if (threadIdx.x == 0) atomicAdd(&sc->T, (double)red[0]);
}

__global__ void k7_initrc(float* r, float* c, const Scal* sc) {
    int i = blockIdx.x * 256 + threadIdx.x;
    if (i < KCODES) r[i] = (float)(1.0 / (sc->T + 1e-8));
    if (i < NPTS)   c[i] = 1.0f;
}

__global__ void k8_zeroS(float* S) { S[blockIdx.x * 256 + threadIdx.x] = 0.f; }

// S[k] = sum_n E[n][k] * c[n]  (block covers 128 rows; thread -> 4 strided k)
__global__ __launch_bounds__(256) void k9_row(const float* __restrict__ E,
        const float* __restrict__ c, float* __restrict__ S) {
    const int t = threadIdx.x;
    const int nbase = blockIdx.x * 128;
    float p0 = 0.f, p1 = 0.f, p2 = 0.f, p3 = 0.f;
    for (int rr = 0; rr < 128; ++rr) {
        const int n = nbase + rr;
        const float cn = c[n];
        const float* row = E + (size_t)n * KCODES;
        __builtin_prefetch(row + 4 * KCODES, 0, 1);   // global_prefetch_b8 stream-ahead
        p0 = fmaf(row[t],       cn, p0);
        p1 = fmaf(row[t + 256], cn, p1);
        p2 = fmaf(row[t + 512], cn, p2);
        p3 = fmaf(row[t + 768], cn, p3);
    }
    atomicAdd(&S[t],       p0);
    atomicAdd(&S[t + 256], p1);
    atomicAdd(&S[t + 512], p2);
    atomicAdd(&S[t + 768], p3);
}

__global__ void k10_rup(float* r, const float* S) {
    int k = blockIdx.x * 256 + threadIdx.x;
    float rk = r[k];
    r[k] = rk / ((rk * S[k] + EPSI) * (float)KCODES);
}

// T_n = sum_k E[n][k]*r[k]; fused c update (block per n)
__global__ __launch_bounds__(256) void k11_col(const float* __restrict__ E,
        const float* __restrict__ r, float* __restrict__ c) {
    __shared__ float red[256];
    const int n = blockIdx.x, t = threadIdx.x;
    const float* row = E + (size_t)n * KCODES;
    float s = row[t] * r[t] + row[t + 256] * r[t + 256]
            + row[t + 512] * r[t + 512] + row[t + 768] * r[t + 768];
    red[t] = s; __syncthreads();
    for (int st = 128; st > 0; st >>= 1) { if (t < st) red[t] += red[t + st]; __syncthreads(); }
    if (t == 0) {
        float cn = c[n], Tn = red[0];
        c[n] = cn / ((cn * Tn + EPSI) * (float)NPTS);
    }
}

// indices[n] = argmax_k E[n][k]*r[k]  (first max, matching jnp.argmax)
__global__ __launch_bounds__(256) void k12_argmax(const float* __restrict__ E,
        const float* __restrict__ r, int* __restrict__ idx, float* __restrict__ outIdx) {
    __shared__ float sv[256]; __shared__ int si[256];
    const int n = blockIdx.x, t = threadIdx.x;
    const float* row = E + (size_t)n * KCODES;
    float best = -1.0f; int bi = t;
    #pragma unroll
    for (int j = 0; j < 4; ++j) {
        int k = t + j * 256;
        float v = row[k] * r[k];
        if (v > best) { best = v; bi = k; }   // j ascending -> first max per thread
    }
    sv[t] = best; si[t] = bi; __syncthreads();
    for (int st = 128; st > 0; st >>= 1) {
        if (t < st) {
            if (sv[t + st] > sv[t] || (sv[t + st] == sv[t] && si[t + st] < si[t])) {
                sv[t] = sv[t + st]; si[t] = si[t + st];
            }
        }
        __syncthreads();
    }
    if (t == 0) { idx[n] = si[0]; outIdx[n] = (float)si[0]; }
}

// z_q (B,C,H,W) = emb[idx], + MSE accumulation
__global__ __launch_bounds__(256) void k13_gather(const float* __restrict__ z,
        const float* __restrict__ emb, const int* __restrict__ idx,
        float* __restrict__ out, Scal* sc) {
    __shared__ float red[256];
    const size_t i = (size_t)blockIdx.x * 256 + threadIdx.x;
    const int w = (int)(i & 31), h = (int)((i >> 5) & 31);
    const int cc = (int)((i >> 10) & 255), b = (int)(i >> 18);
    const int n = (b << 10) + (h << 5) + w;
    const float e  = emb[(size_t)idx[n] * DDIM + cc];
    const float zv = z[i];
    out[i] = e;                      // straight-through forward == gather
    const float df = e - zv;
    red[threadIdx.x] = df * df; __syncthreads();
    for (int st = 128; st > 0; st >>= 1) { if (threadIdx.x < st) red[threadIdx.x] += red[threadIdx.x + st]; __syncthreads(); }
    if (threadIdx.x == 0) atomicAdd(&sc->mse, (double)red[0]);
}

__global__ void k14_loss(const Scal* sc, float* out) {
    double mse = sc->mse / 8388608.0;
    out[8388608] = (float)(1.25 * mse); // loss
    out[8388609] = (float)(0.25 * mse); // commitment_loss
    out[8388610] = (float)mse;          // codebook_loss
}

// ---------------------------------------------------------------------------

extern "C" void kernel_launch(void* const* d_in, const int* in_sizes, int n_in,
                              void* d_out, int out_size, void* d_ws, size_t ws_size,
                              hipStream_t stream) {
    const float* z    = (const float*)d_in[0];
    const float* E0   = (const float*)d_in[1];
    const float* Wm   = (const float*)d_in[2];
    const float* bias = (const float*)d_in[3];
    float* out = (float*)d_out;

    // workspace layout (~154 MB total)
    char* w = (char*)d_ws;
    size_t off = 0;
    Scal*  sc   = (Scal*)(w + off);        off += 256;
    float* y2   = (float*)(w + off);       off += KCODES * 4;
    float* rv   = (float*)(w + off);       off += KCODES * 4;
    float* S    = (float*)(w + off);       off += KCODES * 4;
    int*   idx  = (int*)(w + off);         off += NPTS * 4;
    float* x2   = (float*)(w + off);       off += NPTS * 4;
    float* cv   = (float*)(w + off);       off += NPTS * 4;
    float* emb  = (float*)(w + off);       off += (size_t)KCODES * DDIM * 4;
    __bf16* embB = (__bf16*)(w + off);     off += (size_t)KCODES * DDIM * 2;
    __bf16* zB   = (__bf16*)(w + off);     off += (size_t)NPTS * DDIM * 2;
    float* Ebuf  = (float*)(w + off);      off += (size_t)NPTS * KCODES * 4;  // d, then E in-place

    k0_init   <<<1, 256, 0, stream>>>(sc);
    k1_linear <<<KCODES, 256, 0, stream>>>(E0, Wm, bias, emb);
    k2_embprep<<<KCODES, 256, 0, stream>>>(emb, embB, y2);
    k3_zprep  <<<NPTS, 256, 0, stream>>>(z, zB, x2);

    k4_gemm   <<<dim3(NPTS / 128, KCODES / 128), 256, 0, stream>>>(zB, embB, x2, y2, Ebuf, sc);

    k5_coeff  <<<1, 1, 0, stream>>>(sc);
    k6_exp    <<<NPTS * KCODES / 1024, 256, 0, stream>>>(Ebuf, sc);
    k7_initrc <<<NPTS / 256, 256, 0, stream>>>(rv, cv, sc);

    for (int it = 0; it < 5; ++it) {
        k8_zeroS <<<KCODES / 256, 256, 0, stream>>>(S);
        k9_row   <<<NPTS / 128, 256, 0, stream>>>(Ebuf, cv, S);
        k10_rup  <<<KCODES / 256, 256, 0, stream>>>(rv, S);
        k11_col  <<<NPTS, 256, 0, stream>>>(Ebuf, rv, cv);
    }

    k12_argmax<<<NPTS, 256, 0, stream>>>(Ebuf, rv, idx, out + 8388611);
    k13_gather<<<8388608 / 256, 256, 0, stream>>>(z, emb, idx, out, sc);
    k14_loss  <<<1, 1, 0, stream>>>(sc, out);
}
